// DyadicCrossAttention_30691836297755
// MI455X (gfx1250) — matrix operand order repaired and checked by reading, not compiled, not run on hardware
//
#include <hip/hip_runtime.h>

// ---------------------------------------------------------------------------
// DyadicCrossAttention for MI455X (gfx1250, wave32, WMMA 16x16x32 f16).
//   Stage 1: 6x projection GEMMs (TDM tensor_load_to_lds staging -> LDS,
//            WMMA f16 in / f32 acc; scale folded into Q).
//   Stage 2: 2x flash cross-attention, 2 query tiles per wave:
//            S^T = K^T Q (WMMA), lane-pair softmax via xor-16 bpermute,
//            P^T assembled in registers, O^T += V P^T (WMMA).
//   Stage 3: 2x output-projection GEMMs -> f32 d_out.
// ---------------------------------------------------------------------------

#ifndef USE_TDM
#define USE_TDM 1   // set to 0 to fall back to the manual LDS staging loop
#endif

typedef __attribute__((ext_vector_type(16))) _Float16 v16h;
typedef __attribute__((ext_vector_type(8)))  _Float16 v8h;
typedef __attribute__((ext_vector_type(8)))  float    v8f;
typedef __attribute__((ext_vector_type(4)))  unsigned tdm_u4;
typedef __attribute__((ext_vector_type(8)))  int      tdm_i8;
typedef __attribute__((ext_vector_type(4)))  int      tdm_i4;

static constexpr int   BN    = 2;
static constexpr int   DIM1  = 256;
static constexpr int   DIM2  = 512;
static constexpr int   HID   = 256;
static constexpr int   HEADS = 8;
static constexpr int   HD    = 32;          // head dim == WMMA K exactly
static constexpr int   NPIX  = 48 * 48;     // 2304, divisible by 16/32/64/128
static constexpr int   OUTC  = 256;
static constexpr float SCALE = 0.17677669529663687f;  // HD^-0.5

#define DEVI __device__ __forceinline__

DEVI v8f wmma_f16(v16h a, v16h b, v8f c) {
  return __builtin_amdgcn_wmma_f32_16x16x32_f16(false, a, false, b,
                                                (short)0, c, false, false);
}

DEVI v16h cat8(v8h lo, v8h hi) {
  v16h r;
#pragma unroll
  for (int i = 0; i < 8; ++i) { r[i] = lo[i]; r[i + 8] = hi[i]; }
  return r;
}

DEVI unsigned pack2h(float a, float b) {
  union { _Float16 h[2]; unsigned u; } x;
  x.h[0] = (_Float16)a; x.h[1] = (_Float16)b;
  return x.u;
}

// ---------------------------------------------------------------------------
// Projection GEMM: Out[b,m,n] = sum_c W[m,c] * X[b,c,n] + bias[m]
//   OUTMODE 0 = f32 [B,256,N]; 1 = f16 [B,HID,N] (V); 2 = f16 [B,HEADS,N,HD]
// One wave computes a 16(M)x16(N) tile; block = 4 waves = 16x64 tile.
// X tile (32K x 64N) staged into LDS row-major [k][64] by the Tensor Data
// Mover (D# built per CDNA5 ISA 8.3/8.4), fenced with s_wait_tensorcnt.
// ---------------------------------------------------------------------------
template <int K, typename XT, int OUTMODE, bool SCALEQ>
__global__ void __launch_bounds__(128)
proj_kernel(const float* __restrict__ W, const XT* __restrict__ X,
            const float* __restrict__ bias, void* __restrict__ Out) {
  __shared__ __align__(16) XT xs[32 * 64];   // [k][pixel], raw element type

  const int  tid  = threadIdx.x;
  const int  lane = tid & 31;
  const int  wave = tid >> 5;
  const int  l15  = lane & 15;
  const bool hi   = lane >= 16;
  const int  nblk = blockIdx.x * 64;
  const int  m0   = blockIdx.y * 16;
  const int  b    = blockIdx.z;

  const float* wrow = W + (size_t)(m0 + l15) * K;
  const int pix = wave * 16 + l15;

  v8f acc;
#pragma unroll
  for (int r = 0; r < 8; ++r) acc[r] = 0.f;

  for (int k0 = 0; k0 < K; k0 += 32) {
    __syncthreads();  // previous-iteration readers done before overwrite

#if USE_TDM && defined(__HIP_DEVICE_COMPILE__) && \
    __has_builtin(__builtin_amdgcn_tensor_load_to_lds)
    if (wave == 0) {
      const XT* gsrc = X + ((size_t)b * K + k0) * NPIX + nblk;
      const unsigned long long ga = (unsigned long long)(size_t)gsrc;
      const unsigned lds = (unsigned)(size_t)(&xs[0]);
      // D# group 0: count=1, lds_addr, global_addr[56:0], type=2
      tdm_u4 g0 = { 1u, lds, (unsigned)ga,
                    (unsigned)((ga >> 32) & 0x01FFFFFFull) | 0x80000000u };
      // D# group 1: data_size, tensor_dim0/1, tile_dim0/1, dim0 stride
      constexpr int DSZ = (sizeof(XT) == 4) ? 2 : 1;   // 4B / 2B elements
      tdm_i8 g1 = { DSZ << 16,                         // wg_mask=0, data_size
                    (int)((NPIX & 0xFFFF) << 16),      // tensor_dim0 lo16
                    (int)((K & 0xFFFF) << 16),         // dim0 hi=0, tensor_dim1 lo16
                    64 << 16,                          // dim1 hi=0, tile_dim0=64
                    32,                                // tile_dim1=32
                    NPIX,                              // tensor_dim0_stride lo32
                    0, 0 };
      tdm_i4 g2 = { 0, 0, 0, 0 };
      tdm_i4 g3 = { 0, 0, 0, 0 };
#if __clang_major__ >= 23
      tdm_i8 g4 = { 0, 0, 0, 0, 0, 0, 0, 0 };
      __builtin_amdgcn_tensor_load_to_lds(g0, g1, g2, g3, g4, 0);
#else
      __builtin_amdgcn_tensor_load_to_lds(g0, g1, g2, g3, 0);
#endif
      __builtin_amdgcn_s_wait_tensorcnt(0);
    }
#else
    // Fallback: cooperative staging in the same [k][64] layout.
#pragma unroll 4
    for (int i = tid; i < 32 * 64; i += 128) {
      const int kk = i >> 6, px = i & 63;
      xs[kk * 64 + px] = X[((size_t)b * K + k0 + kk) * NPIX + nblk + px];
    }
    if (k0 + 32 < K)
      __builtin_prefetch(&X[((size_t)b * K + k0 + 32) * NPIX + nblk + (tid & 63)], 0, 0);
#endif
    __syncthreads();

    // A operand (W tile, 16x32): lanes 0-15 hold K {0-7,16-23}, 16-31 {8-15,24-31}
    const float* wp = wrow + k0 + (hi ? 8 : 0);
    const float4 f0 = *(const float4*)(wp + 0);
    const float4 f1 = *(const float4*)(wp + 4);
    const float4 f2 = *(const float4*)(wp + 16);
    const float4 f3 = *(const float4*)(wp + 20);
    v16h a;
    a[0]  = (_Float16)f0.x; a[1]  = (_Float16)f0.y; a[2]  = (_Float16)f0.z; a[3]  = (_Float16)f0.w;
    a[4]  = (_Float16)f1.x; a[5]  = (_Float16)f1.y; a[6]  = (_Float16)f1.z; a[7]  = (_Float16)f1.w;
    a[8]  = (_Float16)f2.x; a[9]  = (_Float16)f2.y; a[10] = (_Float16)f2.z; a[11] = (_Float16)f2.w;
    a[12] = (_Float16)f3.x; a[13] = (_Float16)f3.y; a[14] = (_Float16)f3.z; a[15] = (_Float16)f3.w;

    // B operand (X tile, 32x16): lane = pixel col; element e = k row.
    v16h bv;
    const int kb = hi ? 8 : 0;
#pragma unroll
    for (int e = 0; e < 16; ++e)
      bv[e] = (_Float16)(float)xs[(kb + e + (e & 8)) * 64 + pix];

    acc = wmma_f16(a, bv, acc);
  }

  const int ncol  = nblk + pix - wave * 16 + wave * 16;  // == nblk + pix
  const int rbase = hi ? 8 : 0;
#pragma unroll
  for (int r = 0; r < 8; ++r) {
    const int M = m0 + rbase + r;
    float v = acc[r] + bias[M];
    if (SCALEQ) v *= SCALE;
    if (OUTMODE == 0) {
      ((float*)Out)[((size_t)b * OUTC + M) * NPIX + ncol] = v;
    } else if (OUTMODE == 1) {
      ((_Float16*)Out)[((size_t)b * HID + M) * NPIX + ncol] = (_Float16)v;
    } else {
      const int hh = M >> 5, d = M & 31;
      ((_Float16*)Out)[(((size_t)(b * HEADS + hh)) * NPIX + ncol) * HD + d] = (_Float16)v;
    }
  }
}

// ---------------------------------------------------------------------------
// Flash cross-attention, TWO 16-row query tiles per wave (K/V tiles loaded
// once feed 8 WMMAs -> 2x arithmetic intensity on streamed K/V, and two
// independent S/softmax chains hide WMMA latency).
//   Qt, Kt : f16 [B, HEADS, N, HD]  (d-contiguous; Q pre-scaled by HD^-0.5)
//   Vc     : f16 [B, HID, N]        (m-contiguous)
//   Oc     : f16 [B, HID, N]        attention output, channel-major
// ---------------------------------------------------------------------------
__global__ void __launch_bounds__(128)
attn_kernel(const _Float16* __restrict__ Qt, const _Float16* __restrict__ Kt,
            const _Float16* __restrict__ Vc, _Float16* __restrict__ Oc) {
  const int  lane = threadIdx.x & 31;
  const int  wave = threadIdx.x >> 5;
  const int  l15  = lane & 15;
  const bool hi   = lane >= 16;
  const int  h    = blockIdx.y;
  const int  b    = blockIdx.z;
  const int  nbase = (blockIdx.x * 4 + wave) * 32;

  const size_t headOff = (size_t)(b * HEADS + h) * NPIX * HD;

  // Hoisted B operands for S^T: Q tiles (32d x 16n), lanes 0-15 hold d {0-7,16-23}, 16-31 {8-15,24-31}.
  v16h qB[2];
#pragma unroll
  for (int t = 0; t < 2; ++t) {
    const _Float16* qp = Qt + headOff + (size_t)(nbase + t * 16 + l15) * HD + (hi ? 8 : 0);
    qB[t] = cat8(*(const v8h*)qp, *(const v8h*)(qp + 16));
  }

  const _Float16* kbase = Kt + headOff;
  const _Float16* v0p = Vc + (size_t)(b * HID + h * HD + l15) * NPIX;  // d = l15
  const _Float16* v1p = v0p + (size_t)16 * NPIX;                       // d = 16 + l15

  v8f o0[2], o1[2], zc;
#pragma unroll
  for (int r = 0; r < 8; ++r) { o0[0][r] = o0[1][r] = o1[0][r] = o1[1][r] = 0.f; zc[r] = 0.f; }
  float mrun[2] = { -1e30f, -1e30f };
  float lrun[2] = { 0.f, 0.f };

  for (int mg = 0; mg < NPIX; mg += 32) {
    // A operands shared by both query tiles: K rows m, V rows d.
    const _Float16* kp0 = kbase + (size_t)(mg + l15) * HD + (hi ? 8 : 0);
    const _Float16* kp1 = kp0 + (size_t)16 * HD;
    const v16h aK0 = cat8(*(const v8h*)kp0, *(const v8h*)(kp0 + 16));
    const v16h aK1 = cat8(*(const v8h*)kp1, *(const v8h*)(kp1 + 16));
    const int co = hi ? 16 : 0;
    const v16h aV0 = cat8(*(const v8h*)(v0p + mg + co), *(const v8h*)(v0p + mg + co + 8));
    const v16h aV1 = cat8(*(const v8h*)(v1p + mg + co), *(const v8h*)(v1p + mg + co + 8));

    v16h bp[2];
#pragma unroll
    for (int t = 0; t < 2; ++t) {
      // S^T tiles: lane l<16 holds (m = mg{+16}+r, n = nbase+16t+l); l+16 -> m+8.
      const v8f s0 = wmma_f16(aK0, qB[t], zc);
      const v8f s1 = wmma_f16(aK1, qB[t], zc);

      // Online softmax: each row n is exactly the lane pair (l, l+16).
      float mx = -1e30f;
#pragma unroll
      for (int r = 0; r < 8; ++r) mx = fmaxf(mx, fmaxf(s0[r], s1[r]));
      mx = fmaxf(mx, __shfl_xor(mx, 16, 32));
      const float mnew  = fmaxf(mrun[t], mx);
      const float alpha = __expf(mrun[t] - mnew);
      mrun[t] = mnew;

      float p0[8], p1[8], psum = 0.f;
#pragma unroll
      for (int r = 0; r < 8; ++r) {
        p0[r] = __expf(s0[r] - mnew);
        p1[r] = __expf(s1[r] - mnew);
        psum += p0[r] + p1[r];
      }
      psum += __shfl_xor(psum, 16, 32);
      lrun[t] = lrun[t] * alpha + psum;
#pragma unroll
      for (int r = 0; r < 8; ++r) { o0[t][r] *= alpha; o1[t][r] *= alpha; }

      // Assemble P^T B-operand (32m x 16n) in registers via xor-16 shuffles.
      unsigned u0[4], u1[4], s0u[4], s1u[4];
#pragma unroll
      for (int j = 0; j < 4; ++j) {
        u0[j] = pack2h(p0[2 * j], p0[2 * j + 1]);
        u1[j] = pack2h(p1[2 * j], p1[2 * j + 1]);
      }
#pragma unroll
      for (int j = 0; j < 4; ++j) {
        s0u[j] = __shfl_xor(u0[j], 16, 32);
        s1u[j] = __shfl_xor(u1[j], 16, 32);
      }
      union { unsigned u[8]; v16h v; } bpu;
#pragma unroll
      for (int j = 0; j < 4; ++j) {
        bpu.u[j]     = hi ? s1u[j] : u0[j];  // m rows mg+0..7  / mg+16..23
        bpu.u[4 + j] = hi ? u1[j]  : s0u[j]; // m rows mg+8..15 / mg+24..31
      }
      bp[t] = bpu.v;
    }

#pragma unroll
    for (int t = 0; t < 2; ++t) {
      o0[t] = wmma_f16(aV0, bp[t], o0[t]);
      o1[t] = wmma_f16(aV1, bp[t], o1[t]);
    }
  }

#pragma unroll
  for (int t = 0; t < 2; ++t) {
    const float inv = 1.0f / lrun[t];  // lane pair shares the row sum
    _Float16* obase = Oc + (size_t)(b * HID + h * HD) * NPIX + (nbase + t * 16 + l15);
#pragma unroll
    for (int r = 0; r < 8; ++r) {
      const int d = (hi ? 8 : 0) + r;
      obase[(size_t)d * NPIX]        = (_Float16)(o0[t][r] * inv);
      obase[(size_t)(16 + d) * NPIX] = (_Float16)(o1[t][r] * inv);
    }
  }
}

// ---------------------------------------------------------------------------
extern "C" void kernel_launch(void* const* d_in, const int* in_sizes, int n_in,
                              void* d_out, int out_size, void* d_ws, size_t ws_size,
                              hipStream_t stream) {
  (void)in_sizes; (void)n_in; (void)out_size; (void)ws_size;
  const float* m1  = (const float*)d_in[0];
  const float* m2  = (const float*)d_in[1];
  const float* Wq1 = (const float*)d_in[2];  const float* bq1 = (const float*)d_in[3];
  const float* Wk1 = (const float*)d_in[4];  const float* bk1 = (const float*)d_in[5];
  const float* Wv1 = (const float*)d_in[6];  const float* bv1 = (const float*)d_in[7];
  const float* Wq2 = (const float*)d_in[8];  const float* bq2 = (const float*)d_in[9];
  const float* Wk2 = (const float*)d_in[10]; const float* bk2 = (const float*)d_in[11];
  const float* Wv2 = (const float*)d_in[12]; const float* bv2 = (const float*)d_in[13];
  const float* Wo1 = (const float*)d_in[14]; const float* bo1 = (const float*)d_in[15];
  const float* Wo2 = (const float*)d_in[16]; const float* bo2 = (const float*)d_in[17];

  _Float16* ws = (_Float16*)d_ws;
  const size_t S = (size_t)BN * HID * NPIX;  // 1,179,648 halves per buffer
  _Float16* Qt1 = ws + 0 * S;  _Float16* Kt1 = ws + 1 * S;  _Float16* V1 = ws + 2 * S;
  _Float16* Qt2 = ws + 3 * S;  _Float16* Kt2 = ws + 4 * S;  _Float16* V2 = ws + 5 * S;
  _Float16* AO1 = ws + 6 * S;  _Float16* AO2 = ws + 7 * S;

  const dim3 blk(128);
  const dim3 gp(NPIX / 64, HID / 16, BN);

  // Stage 1: projections (scale folded into Q; Q/K head-transposed, V natural)
  proj_kernel<DIM1, float, 2, true ><<<gp, blk, 0, stream>>>(Wq1, m1, bq1, Qt1);
  proj_kernel<DIM1, float, 2, false><<<gp, blk, 0, stream>>>(Wk1, m1, bk1, Kt1);
  proj_kernel<DIM1, float, 1, false><<<gp, blk, 0, stream>>>(Wv1, m1, bv1, V1);
  proj_kernel<DIM2, float, 2, true ><<<gp, blk, 0, stream>>>(Wq2, m2, bq2, Qt2);
  proj_kernel<DIM2, float, 2, false><<<gp, blk, 0, stream>>>(Wk2, m2, bk2, Kt2);
  proj_kernel<DIM2, float, 1, false><<<gp, blk, 0, stream>>>(Wv2, m2, bv2, V2);

  // Stage 2: bidirectional flash cross-attention (2 tiles/wave)
  const dim3 ga(NPIX / 128, HEADS, BN);  // 18 blocks * 4 waves * 32 rows = 2304
  attn_kernel<<<ga, blk, 0, stream>>>(Qt1, Kt2, V2, AO1);
  attn_kernel<<<ga, blk, 0, stream>>>(Qt2, Kt1, V1, AO2);

  // Stage 3: output projections -> f32 d_out (output1 then output2)
  float* out1 = (float*)d_out;
  float* out2 = out1 + (size_t)BN * OUTC * NPIX;
  proj_kernel<HID, _Float16, 0, false><<<gp, blk, 0, stream>>>(Wo1, AO1, bo1, out1);
  proj_kernel<HID, _Float16, 0, false><<<gp, blk, 0, stream>>>(Wo2, AO2, bo2, out2);
}
